// GatedLinearAttention2_57518202028658
// MI455X (gfx1250) — compile-verified, matrix operand order checked by reading, not run on hardware
//
#include <hip/hip_runtime.h>
#include <hip/hip_bf16.h>
#include <math.h>

// ---------------------------------------------------------------------------
// Problem constants (from reference): B=2, L=2048, H=2048, K=128
// ---------------------------------------------------------------------------
#define BB 2
#define LL 2048
#define HH 2048
#define KK 128
#define CHUNK 32
#define VW 128
#define GLA_SCALE 0.08838834764831845f /* 128^-0.5 */
#define EPS 1e-5f

#define AS1 __attribute__((address_space(1)))
#define AS3 __attribute__((address_space(3)))

typedef __attribute__((ext_vector_type(16))) __bf16 v16bf;
typedef __attribute__((ext_vector_type(8)))  float v8f;
typedef __attribute__((ext_vector_type(4)))  unsigned int v4u;
typedef __attribute__((ext_vector_type(8)))  int v8i;
typedef __attribute__((ext_vector_type(4)))  int v4i;

#if __has_builtin(__builtin_amdgcn_global_load_async_to_lds_b128) && \
    __has_builtin(__builtin_amdgcn_s_wait_asynccnt)
#define HAS_ASYNC 1
#else
#define HAS_ASYNC 0
#endif

#if __has_builtin(__builtin_amdgcn_tensor_load_to_lds) && \
    __has_builtin(__builtin_amdgcn_s_wait_tensorcnt)
#define HAS_TDM 1
#else
#define HAS_TDM 0
#endif

union BV { unsigned int u[8]; v16bf v; };

__device__ __forceinline__ unsigned short f2bf(float f) {
    unsigned int u = __float_as_uint(f);
    u += 0x7FFFu + ((u >> 16) & 1u);   // round-to-nearest-even
    return (unsigned short)(u >> 16);
}

// Pack a wave32 WMMA 16-bit fragment (A-style: one matrix row per lane,
// K split 0..7 / 8..15 between lane halves, +16 in upper 4 VGPRs).
// Caller passes row = r0 + (lane & 15). Works for global or LDS pointers.
__device__ __forceinline__ v16bf load_frag(const unsigned short* __restrict__ base,
                                           int row, int rs, int kbase, int lane) {
    BV r;
    const unsigned short* p = base + row * rs + kbase + 8 * (lane >> 4);
#pragma unroll
    for (int v = 0; v < 8; ++v) {
        int kidx = 16 * (v >> 2) + 2 * (v & 3);
        r.u[v] = *(const unsigned int*)(p + kidx);
    }
    return r.v;
}

__device__ __forceinline__ v8f wmma_bf16(v16bf a, v16bf b, v8f c) {
    return __builtin_amdgcn_wmma_f32_16x16x32_bf16(false, a, false, b, (short)0, c,
                                                   false, false);
}

// ---------------------------------------------------------------------------
// fp32 -> bf16 conversion
// ---------------------------------------------------------------------------
__global__ void cvt_bf16_kernel(const float* __restrict__ in,
                                unsigned short* __restrict__ out, int n) {
    int i = blockIdx.x * blockDim.x + threadIdx.x;
    if (i < n) out[i] = f2bf(in[i]);
}

// ---------------------------------------------------------------------------
// C[M,N] = A[M,Kd](bf16) @ W[N,Kd](bf16)^T  with fused epilogues.
// block = 256 thr (8 waves) arranged 4(m) x 2(n); each wave owns a 32x32
// region (2x2 WMMA tiles); block tile 128x64. The shared B tile (64 rows x
// 32 k, bf16) is staged into LDS once per K-step, double-buffered with
// GLOBAL_LOAD_ASYNC_TO_LDS_B128 + s_wait_asynccnt when available.
// mode 0: plain   mode 1: silu(v)*SCALE   mode 2: sigmoid -> C, -log1p(exp(.)) -> out2
// mode 3: accumulate onto Cin (== C)
// ---------------------------------------------------------------------------
__device__ __forceinline__ void stage_b(const unsigned short* __restrict__ W,
                                        int Kd, int n0blk, int kk,
                                        unsigned short* Bs, int tid) {
    int r = tid >> 2;            // 64 rows, 4 threads per row
    int cidx = (tid & 3) * 8;    // 8 ushorts (16B) per thread
    const unsigned short* src = W + (n0blk + r) * Kd + kk + cidx;
    unsigned short* dst = Bs + r * 32 + cidx;
#if HAS_ASYNC
    __builtin_amdgcn_global_load_async_to_lds_b128((AS1 v4i*)src, (AS3 v4i*)dst,
                                                   0, 0);
#else
    *(uint4*)dst = *(const uint4*)src;
#endif
}

__global__ void gemm_bf16_kernel(const unsigned short* __restrict__ A,
                                 const unsigned short* __restrict__ W,
                                 float* __restrict__ C,
                                 const float* __restrict__ Cin,
                                 float* __restrict__ out2,
                                 int M, int N, int Kd, int mode) {
    __shared__ unsigned short Bs[2][64 * 32];

    int tid  = threadIdx.x;
    int lane = tid & 31;
    int w    = tid >> 5;
    int half = lane >> 4;
    int l15  = lane & 15;
    int wm = w >> 1, wn = w & 1;
    int n0blk = blockIdx.y * 64;
    int m0w = blockIdx.x * 128 + wm * 32;
    int n0w = wn * 32;

    v8f c[2][2];
    if (mode == 3) {
#pragma unroll
        for (int mt = 0; mt < 2; ++mt)
#pragma unroll
            for (int nt = 0; nt < 2; ++nt)
#pragma unroll
                for (int i = 0; i < 8; ++i)
                    c[mt][nt][i] = Cin[(m0w + mt * 16 + i + 8 * half) * N +
                                       n0blk + n0w + nt * 16 + l15];
    } else {
#pragma unroll
        for (int mt = 0; mt < 2; ++mt)
#pragma unroll
            for (int nt = 0; nt < 2; ++nt)
#pragma unroll
                for (int i = 0; i < 8; ++i) c[mt][nt][i] = 0.f;
    }

    stage_b(W, Kd, n0blk, 0, Bs[0], tid);
    int buf = 0;
    int arow0 = m0w + l15;
    for (int kk = 0; kk < Kd; kk += 32) {
        bool more = (kk + 32) < Kd;
        if (more) stage_b(W, Kd, n0blk, kk + 32, Bs[buf ^ 1], tid);
#if HAS_ASYNC
        if (more) __builtin_amdgcn_s_wait_asynccnt(1);
        else      __builtin_amdgcn_s_wait_asynccnt(0);
#endif
        __syncthreads();
        if (more) {
            __builtin_prefetch(A + arow0 * Kd + kk + 32, 0, 1);
            __builtin_prefetch(A + (arow0 + 16) * Kd + kk + 32, 0, 1);
        }
        v16bf a0 = load_frag(A, arow0,      Kd, kk, lane);
        v16bf a1 = load_frag(A, arow0 + 16, Kd, kk, lane);
        v16bf b0 = load_frag(Bs[buf], n0w + l15,      32, 0, lane);
        v16bf b1 = load_frag(Bs[buf], n0w + 16 + l15, 32, 0, lane);
        c[0][0] = wmma_bf16(a0, b0, c[0][0]);
        c[0][1] = wmma_bf16(a0, b1, c[0][1]);
        c[1][0] = wmma_bf16(a1, b0, c[1][0]);
        c[1][1] = wmma_bf16(a1, b1, c[1][1]);
        buf ^= 1;
        __syncthreads();
    }

#pragma unroll
    for (int mt = 0; mt < 2; ++mt)
#pragma unroll
        for (int nt = 0; nt < 2; ++nt)
#pragma unroll
            for (int i = 0; i < 8; ++i) {
                int m = m0w + mt * 16 + i + 8 * half;
                int n = n0blk + n0w + nt * 16 + l15;
                float v = c[mt][nt][i];
                if (mode == 1) {
                    C[m * N + n] = v / (1.f + __expf(-v)) * GLA_SCALE;
                } else if (mode == 2) {
                    float ksv = 1.f / (1.f + __expf(-v));
                    C[m * N + n] = ksv;
                    out2[m * N + n] = -log1pf(__expf(ksv));   // log_sigmoid(-ksv)
                } else {
                    C[m * N + n] = v;
                }
            }
}

// ---------------------------------------------------------------------------
// TDM: issue a 2D tile load (32 rows x 128 f32, row stride 2048 f32) from
// global x into an LDS staging buffer.  D# built per CDNA5 ISA §8.3-8.4:
//   g0: count=1 | lds_addr | global_addr | type=2
//   g1: data_size=4B, tensor_dim0=128, tensor_dim1=32, tile 128x32,
//       tensor_dim0_stride=2048
// ---------------------------------------------------------------------------
#if HAS_TDM
__device__ __forceinline__ void tdm_load_tile(const float* gsrc, void* lds_dst) {
    unsigned long ga  = (unsigned long)(uintptr_t)gsrc;
    unsigned      lab = (unsigned)(uintptr_t)lds_dst;   // LDS aperture low 32 = offset
    v4u g0 = { 1u, lab, (unsigned)ga,
               (unsigned)((ga >> 32) & 0x01FFFFFFu) | 0x80000000u };
    v8i g1 = { (int)(2u << 16),          // data_size = 4B
               (int)(128u << 16),        // tensor_dim0 = 128 (lo 16)
               (int)(32u << 16),         // tensor_dim0 hi=0 | tensor_dim1 = 32
               (int)(128u << 16),        // tensor_dim1 hi=0 | tile_dim0 = 128
               (int)32,                  // tile_dim1 = 32, tile_dim2 = 0
               (int)2048,                // tensor_dim0_stride lo
               0, 0 };
    v4i z4 = { 0, 0, 0, 0 };
    v8i z8 = { 0, 0, 0, 0, 0, 0, 0, 0 };
    __builtin_amdgcn_tensor_load_to_lds(g0, g1, z4, z4, z8, 0);
}
#endif

// ---------------------------------------------------------------------------
// Chunked GLA scan. grid = B*16 blocks; block owns (batch b, 128-wide V slice),
// keeps state S[128][128] fp32 in LDS, iterates 64 chunks of 32 timesteps.
// All chunk-local contractions are v_wmma_f32_16x16x32_bf16.  The V chunk is
// prefetched one chunk ahead by the Tensor Data Mover (double-buffered LDS).
// ---------------------------------------------------------------------------
__global__ void gla_kernel(const float* __restrict__ q,
                           const float* __restrict__ ksig,
                           const float* __restrict__ g,
                           const float* __restrict__ x,
                           float* __restrict__ o) {
    extern __shared__ char smem[];
    float*          S  = (float*)smem;                               // 128*128 f32
    float*          D  = (float*)(smem + 128 * 128 * 4);             // 32*128 f32
    unsigned short* sT = (unsigned short*)(smem + 128 * 128 * 4 + CHUNK * 128 * 4);
    unsigned short* qe  = sT  + 128 * 128;   // [32][128]
    unsigned short* kf  = qe  + CHUNK * 128; // [32][128]
    unsigned short* kgT = kf  + CHUNK * 128; // [128][32]
    unsigned short* vT  = kgT + 128 * CHUNK; // [128][32]
    unsigned short* Pm  = vT  + 128 * CHUNK; // [32][32]
    float*          xstg = (float*)(Pm + CHUNK * CHUNK); // [2][32*128] f32

    int tid  = threadIdx.x;
    int lane = tid & 31;
    int w    = tid >> 5;
    int half = lane >> 4;
    int l15  = lane & 15;
    int b  = blockIdx.x >> 4;
    int v0 = (blockIdx.x & 15) * VW;
    const int NC = LL / CHUNK;

#if HAS_TDM
    if (w == 0) tdm_load_tile(x + (size_t)(b * LL) * HH + v0, xstg);
#endif
    for (int i = tid; i < 128 * 128; i += 256) S[i] = 0.f;
    __syncthreads();

    for (int cidx = 0; cidx < NC; ++cidx) {
        int l0 = cidx * CHUNK;
        int base = (b * LL + l0) * KK;

#if HAS_TDM
        if (w == 0 && (cidx + 1) < NC)   // prefetch next chunk's V tile
            tdm_load_tile(x + (size_t)(b * LL + l0 + CHUNK) * HH + v0,
                          xstg + ((cidx + 1) & 1) * CHUNK * VW);
#endif
        // 1) cumulative log-decay D[t][k] = sum_{s<=t} g[s][k]
        if (tid < 128) {
            float acc = 0.f;
            for (int t = 0; t < CHUNK; ++t) {
                acc += g[base + t * KK + tid];
                D[t * 128 + tid] = acc;
            }
        }
        __syncthreads();
#if HAS_TDM
        if (w == 0) {
            if ((cidx + 1) < NC) __builtin_amdgcn_s_wait_tensorcnt(1);
            else                 __builtin_amdgcn_s_wait_tensorcnt(0);
        }
        __syncthreads();
        const float* xs = xstg + (cidx & 1) * CHUNK * VW;
#endif

        // 2) decay-weighted bf16 operand tiles
        for (int i = tid; i < CHUNK * 128; i += 256) {
            int t = i >> 7, k = i & 127;
            float d  = D[i];
            float dl = D[(CHUNK - 1) * 128 + k];
            float qv = q[base + t * KK + k];
            float kv = ksig[base + t * KK + k];
            qe[i] = f2bf(qv * __expf(d));                   // Q * exp(D)
            kf[i] = f2bf(kv * __expf(-d));                  // K * exp(-D)
            kgT[k * CHUNK + t] = f2bf(kv * __expf(dl - d)); // K * exp(D_C - D), transposed
        }
        for (int i = tid; i < CHUNK * VW; i += 256) {
            int t = i >> 7, v = i & 127;
#if HAS_TDM
            vT[v * CHUNK + t] = f2bf(xs[t * VW + v]);       // V^T slice (bf16)
#else
            vT[v * CHUNK + t] = f2bf(x[(size_t)(b * LL + l0 + t) * HH + v0 + v]);
#endif
        }
        for (int i = tid; i < 128 * VW; i += 256) {
            int kd = i & 127, v = i >> 7;
            sT[v * 128 + kd] = f2bf(S[kd * 128 + v]);       // S_in^T snapshot (bf16)
        }
        __syncthreads();

        if (w < 4) {
            // 3a) causal P = (Q e^D)(K e^-D)^T : four 16x16 tiles, K-loop 128
            int t0 = (w >> 1) * 16, s0 = (w & 1) * 16;
            v8f c;
#pragma unroll
            for (int i = 0; i < 8; ++i) c[i] = 0.f;
            for (int kk = 0; kk < 128; kk += 32) {
                v16bf a  = load_frag(qe, t0 + l15, 128, kk, lane);
                v16bf bm = load_frag(kf, s0 + l15, 128, kk, lane);
                c = wmma_bf16(a, bm, c);
            }
#pragma unroll
            for (int i = 0; i < 8; ++i) {
                int t = t0 + i + 8 * half, s = s0 + l15;
                Pm[t * CHUNK + s] = f2bf((s <= t) ? c[i] : 0.f);
            }
        } else {
            // 3b) state update S = e^{D_C} (.) S + kg^T V : 64 tiles over waves 4..7
            for (int j = 0; j < 16; ++j) {
                int tt = (w - 4) * 16 + j;
                int m0 = (tt >> 3) * 16, n0 = (tt & 7) * 16;
                v8f c;
#pragma unroll
                for (int i = 0; i < 8; ++i) {
                    int m = m0 + i + 8 * half;
                    c[i] = __expf(D[(CHUNK - 1) * 128 + m]) * S[m * 128 + n0 + l15];
                }
                v16bf a  = load_frag(kgT, m0 + l15, CHUNK, 0, lane);
                v16bf bm = load_frag(vT,  n0 + l15, CHUNK, 0, lane);
                c = wmma_bf16(a, bm, c);
#pragma unroll
                for (int i = 0; i < 8; ++i) {
                    int m = m0 + i + 8 * half;
                    S[m * 128 + n0 + l15] = c[i];
                }
            }
        }
        __syncthreads();

        // 4) O = P V + (Q e^D) S_in : 16 tiles over 8 waves (2 each)
        for (int j = 0; j < 2; ++j) {
            int id = w + j * 8;
            int t0 = (id >> 3) * 16, n0 = (id & 7) * 16;
            v8f c;
#pragma unroll
            for (int i = 0; i < 8; ++i) c[i] = 0.f;
            {
                v16bf a  = load_frag(Pm, t0 + l15, CHUNK, 0, lane);
                v16bf bm = load_frag(vT, n0 + l15, CHUNK, 0, lane);
                c = wmma_bf16(a, bm, c);
            }
            for (int kk = 0; kk < 128; kk += 32) {
                v16bf a  = load_frag(qe, t0 + l15, 128, kk, lane);
                v16bf bm = load_frag(sT, n0 + l15, 128, kk, lane);
                c = wmma_bf16(a, bm, c);
            }
#pragma unroll
            for (int i = 0; i < 8; ++i) {
                int t = t0 + i + 8 * half;
                o[(b * LL + l0 + t) * HH + v0 + n0 + l15] = c[i];
            }
        }
        __syncthreads();
    }
}

// ---------------------------------------------------------------------------
// per-row RMS stats + bf16(o)
// ---------------------------------------------------------------------------
__global__ void rms_kernel(const float* __restrict__ o,
                           unsigned short* __restrict__ ob,
                           float* __restrict__ inv_rms) {
    __shared__ float red[256];
    int row = blockIdx.x;
    const float* p = o + row * HH;
    float s = 0.f;
    for (int i = threadIdx.x; i < HH; i += 256) {
        float v = p[i];
        s += v * v;
        ob[row * HH + i] = f2bf(v);
    }
    red[threadIdx.x] = s;
    __syncthreads();
    for (int d = 128; d > 0; d >>= 1) {
        if (threadIdx.x < d) red[threadIdx.x] += red[threadIdx.x + d];
        __syncthreads();
    }
    if (threadIdx.x == 0) inv_rms[row] = rsqrtf(red[0] / (float)HH + EPS);
}

// ---------------------------------------------------------------------------
// out = rmsnorm(o)*gw * go * sigmoid(go)
// ---------------------------------------------------------------------------
__global__ void final_kernel(const float* __restrict__ o,
                             const float* __restrict__ go,
                             const float* __restrict__ inv_rms,
                             const float* __restrict__ gw,
                             float* __restrict__ out, int n) {
    int i = blockIdx.x * blockDim.x + threadIdx.x;
    if (i < n) {
        int row = i >> 11;       // /H
        int h   = i & (HH - 1);
        float r  = o[i] * inv_rms[row] * gw[h];
        float gv = go[i];
        out[i] = r * gv * (1.f / (1.f + __expf(-gv)));
    }
}

// ---------------------------------------------------------------------------
// host launcher
// ---------------------------------------------------------------------------
extern "C" void kernel_launch(void* const* d_in, const int* in_sizes, int n_in,
                              void* d_out, int out_size, void* d_ws, size_t ws_size,
                              hipStream_t stream) {
    (void)in_sizes; (void)n_in; (void)out_size; (void)ws_size;
    const float* x   = (const float*)d_in[0];
    const float* Wq  = (const float*)d_in[1];
    const float* Wk  = (const float*)d_in[2];
    const float* Wog = (const float*)d_in[3];
    const float* Wig = (const float*)d_in[4];
    const float* gw  = (const float*)d_in[5];
    float* out = (float*)d_out;

    char* ws = (char*)d_ws;
    size_t off = 0;
    auto carve = [&](size_t bytes) {
        size_t o = off;
        off = (off + bytes + 255) & ~(size_t)255;
        return o;
    };
    const int MROWS = BB * LL;                       // 4096
    unsigned short* xb   = (unsigned short*)(ws + carve((size_t)MROWS * HH * 2));
    unsigned short* ob   = (unsigned short*)(ws + carve((size_t)MROWS * HH * 2));
    unsigned short* Wqb  = (unsigned short*)(ws + carve((size_t)KK * HH * 2));
    unsigned short* Wkb  = (unsigned short*)(ws + carve((size_t)KK * HH * 2));
    unsigned short* Wogb = (unsigned short*)(ws + carve((size_t)HH * HH * 2));
    unsigned short* Wigb = (unsigned short*)(ws + carve((size_t)HH * HH * 2));
    float* qbuf  = (float*)(ws + carve((size_t)MROWS * KK * 4));
    float* kbuf  = (float*)(ws + carve((size_t)MROWS * KK * 4));
    float* gbuf  = (float*)(ws + carve((size_t)MROWS * KK * 4));
    float* obuf  = (float*)(ws + carve((size_t)MROWS * HH * 4));
    float* gobuf = (float*)(ws + carve((size_t)MROWS * HH * 4));
    float* irms  = (float*)(ws + carve((size_t)MROWS * 4));

    auto cvt = [&](const float* in, unsigned short* o_, int n) {
        cvt_bf16_kernel<<<(n + 255) / 256, 256, 0, stream>>>(in, o_, n);
    };
    cvt(x,   xb,   MROWS * HH);
    cvt(Wq,  Wqb,  KK * HH);
    cvt(Wk,  Wkb,  KK * HH);
    cvt(Wog, Wogb, HH * HH);
    cvt(Wig, Wigb, HH * HH);

    // q/k projections with fused activations (N=128 -> grid.y = 2)
    gemm_bf16_kernel<<<dim3(MROWS / 128, KK / 64), 256, 0, stream>>>(
        xb, Wqb, qbuf, nullptr, nullptr, MROWS, KK, HH, /*mode=*/1);
    gemm_bf16_kernel<<<dim3(MROWS / 128, KK / 64), 256, 0, stream>>>(
        xb, Wkb, kbuf, nullptr, gbuf, MROWS, KK, HH, /*mode=*/2);

    // chunked GLA scan: 32 blocks (B * 16 V-slices), ~178 KB LDS each
    size_t gla_smem = (size_t)(128 * 128 * 4 + CHUNK * 128 * 4) +
                      (size_t)(128 * 128 + 3 * CHUNK * 128 + 128 * CHUNK + CHUNK * CHUNK) * 2 +
                      (size_t)2 * CHUNK * VW * 4;
    gla_kernel<<<dim3(BB * 16), 256, gla_smem, stream>>>(qbuf, kbuf, gbuf, x, obuf);

    // RMS stats + bf16(o)
    rms_kernel<<<dim3(MROWS), 256, 0, stream>>>(obuf, ob, irms);

    // go = o @ Wog^T + x @ Wig^T
    gemm_bf16_kernel<<<dim3(MROWS / 128, HH / 64), 256, 0, stream>>>(
        ob, Wogb, gobuf, nullptr, nullptr, MROWS, HH, HH, /*mode=*/0);
    gemm_bf16_kernel<<<dim3(MROWS / 128, HH / 64), 256, 0, stream>>>(
        xb, Wigb, gobuf, gobuf, nullptr, MROWS, HH, HH, /*mode=*/3);

    // fused norm/gate epilogue
    int n = MROWS * HH;
    final_kernel<<<(n + 255) / 256, 256, 0, stream>>>(obuf, gobuf, irms, gw, out, n);
}
